// MultiHeadSelfAttention_26259430048382
// MI455X (gfx1250) — matrix operand order repaired and checked by reading, not compiled, not run on hardware
//
#include <hip/hip_runtime.h>

typedef __attribute__((ext_vector_type(16))) _Float16 v16h;
typedef __attribute__((ext_vector_type(8)))  float    v8f;

#define SEQ      4096
#define DMODEL   2048
#define NHEADS   16
#define HDIM     128
#define EPSV     1e-6f

#define WMMA_F16(a, b, c) \
  __builtin_amdgcn_wmma_f32_16x16x32_f16(false, (a), false, (b), (short)0, (c), false, false)

// CDNA5 async global->LDS path (ASYNCcnt-tracked), with sync fallback.
#if defined(__AMDGCN__) && __has_builtin(__builtin_amdgcn_global_load_async_to_lds_b128) && \
    __has_builtin(__builtin_amdgcn_s_wait_asynccnt)
#define HAVE_ASYNC_LDS 1
#else
#define HAVE_ASYNC_LDS 0
#endif

#if HAVE_ASYNC_LDS
typedef int v4i_vs __attribute__((vector_size(16)));
typedef __attribute__((address_space(1))) v4i_vs* async_gptr;
typedef __attribute__((address_space(3))) v4i_vs* async_lptr;
#endif

// Copy 16 bytes (8 halves) global -> LDS.
__device__ inline void async_copy16(const _Float16* gsrc, _Float16* ldst) {
#if HAVE_ASYNC_LDS
  __builtin_amdgcn_global_load_async_to_lds_b128((async_gptr)gsrc, (async_lptr)ldst, 0, 0);
#else
  *(uint4*)ldst = *(const uint4*)gsrc;
#endif
}
__device__ inline void async_wait() {
#if HAVE_ASYNC_LDS
  __builtin_amdgcn_s_wait_asynccnt(0);
#endif
}

// A-fragment (16x32 f16): lane m = L&15, kbase = (L>=16)?8:0.
// halves h: K = 2*kbase + h  -> one contiguous 16-half load.
__device__ inline v16h load_frag_split(const _Float16* rowptr, int kb) {
  union { v16h v; uint4 u[2]; } f;
  f.u[0] = *(const uint4*)(rowptr + 2 * kb);
  f.u[1] = *(const uint4*)(rowptr + 2 * kb + 8);
  return f.v;
}

// B-fragment (32x16 f16): lane n = L&15, K = 16*(L>=16) + h -> one contiguous 16-half load.
__device__ inline v16h load_frag_contig(const _Float16* p) {
  union { v16h v; uint4 u[2]; } f;
  f.u[0] = *(const uint4*)(p);
  f.u[1] = *(const uint4*)(p + 8);
  return f.v;
}

// ---------------------------------------------------------------- convert
__global__ void cvt_f32_f16_kernel(const float* __restrict__ in,
                                   _Float16* __restrict__ out, int n) {
  int i = blockIdx.x * blockDim.x + threadIdx.x;
  int stride = gridDim.x * blockDim.x;
  for (; i < n; i += stride) out[i] = (_Float16)in[i];
}

// ---------------------------------------------------------------- GEMM  y = A * W^T
// A: f16 [M=4096][K=2048] row-major, W: f16 [N=2048][K=2048] row-major (so B = W^T).
// 128x128 tile/block, K-slabs of 64, double-buffered LDS filled with async-to-LDS.
// mode 0: f32 [M][N] to outF.
// mode 1: per-head f16 [head][M][128] to outH (+ optional fused RMSNorm over head dim).
__global__ __launch_bounds__(256)
void gemm_kernel(const _Float16* __restrict__ A, const _Float16* __restrict__ W,
                 float* __restrict__ outF, _Float16* __restrict__ outH,
                 const float* __restrict__ nw, int mode) {
  const int n0   = blockIdx.x * 128;
  const int m0   = blockIdx.y * 128;
  const int tid  = threadIdx.x;
  const int wave = tid >> 5;
  const int lane = tid & 31;
  const int ln   = lane & 15;
  const int hi   = lane >> 4;

  __shared__ alignas(16) _Float16 At[2][128][72];  // 128 rows x 64 k (+8 pad), x2 buffers
  __shared__ alignas(16) _Float16 Bt[2][128][72];

  const int ld_row = tid >> 1;
  const int ld_col = (tid & 1) * 32;

  auto stage_load = [&](int buf, int k0) {
    const _Float16* ga = A + (size_t)(m0 + ld_row) * DMODEL + k0 + ld_col;
    const _Float16* gb = W + (size_t)(n0 + ld_row) * DMODEL + k0 + ld_col;
    _Float16* la = &At[buf][ld_row][ld_col];
    _Float16* lb = &Bt[buf][ld_row][ld_col];
#pragma unroll
    for (int i = 0; i < 4; ++i) {
      async_copy16(ga + i * 8, la + i * 8);
      async_copy16(gb + i * 8, lb + i * 8);
    }
  };

  stage_load(0, 0);
  v8f acc[8] = {};

  for (int k0 = 0; k0 < DMODEL; k0 += 64) {
    const int cur = (k0 >> 6) & 1;
    async_wait();        // my async fills for slab `cur` have landed in LDS
    __syncthreads();     // everyone's fills landed; prior slab's reads all consumed
    if (k0 + 64 < DMODEL) stage_load(cur ^ 1, k0 + 64);

#pragma unroll
    for (int c = 0; c < 2; ++c) {
      v16h a = load_frag_split(&At[cur][wave * 16 + ln][c * 32], hi * 8);
      v16h b8[8];
#pragma unroll
      for (int nf = 0; nf < 8; ++nf)
        b8[nf] = load_frag_contig(&Bt[cur][nf * 16 + ln][c * 32 + hi * 16]);
#pragma unroll
      for (int nf = 0; nf < 8; ++nf) acc[nf] = WMMA_F16(a, b8[nf], acc[nf]);
    }
  }

  if (mode == 0) {
#pragma unroll
    for (int nf = 0; nf < 8; ++nf)
#pragma unroll
      for (int r = 0; r < 8; ++r) {
        int row = m0 + wave * 16 + hi * 8 + r;
        int col = n0 + nf * 16 + ln;
        outF[(size_t)row * DMODEL + col] = acc[nf][r];
      }
  } else {
    // C-frag row mapping: VGPR r <-> row (r + 8*hi); col = nf*16 + ln (within head).
    float g[8];
#pragma unroll
    for (int nf = 0; nf < 8; ++nf) g[nf] = nw ? nw[nf * 16 + ln] : 1.0f;
    float inv[8];
#pragma unroll
    for (int r = 0; r < 8; ++r) {
      if (nw) {
        float ss = 0.f;
#pragma unroll
        for (int nf = 0; nf < 8; ++nf) ss += acc[nf][r] * acc[nf][r];
#pragma unroll
        for (int mask = 1; mask <= 8; mask <<= 1) ss += __shfl_xor(ss, mask, 32);
        inv[r] = rsqrtf(ss * (1.0f / HDIM) + EPSV);
      } else {
        inv[r] = 1.0f;
      }
    }
    const int head = blockIdx.x;
#pragma unroll
    for (int nf = 0; nf < 8; ++nf)
#pragma unroll
      for (int r = 0; r < 8; ++r) {
        int row = m0 + wave * 16 + hi * 8 + r;
        outH[((size_t)head * SEQ + row) * HDIM + nf * 16 + ln] =
            (_Float16)(acc[nf][r] * inv[r] * g[nf]);
      }
  }
}

// ---------------------------------------------------------------- flash attention
// Block = (q-tile of 128 rows, head). Wave w owns q rows w*16..w*16+15.
// Streams K/V in 64-key tiles through LDS; never materializes S x S.
__global__ __launch_bounds__(256)
void flash_attn_kernel(const _Float16* __restrict__ Qh, const _Float16* __restrict__ Kh,
                       const _Float16* __restrict__ Vh, _Float16* __restrict__ Oh) {
  const int h    = blockIdx.y;
  const int q0   = blockIdx.x * 128;
  const int tid  = threadIdx.x;
  const int wave = tid >> 5;
  const int lane = tid & 31;
  const int ln   = lane & 15;
  const int hi   = lane >> 4;

  __shared__ alignas(16) _Float16 Kt[64][HDIM + 8];   // [key][d]   (B-frags contiguous in d)
  __shared__ alignas(16) _Float16 Vt[HDIM][64 + 8];   // [d][key]   (B-frags contiguous in key)
  __shared__ alignas(16) _Float16 Pt[8][16][64 + 8];  // per-wave P tile, C->A relayout

  const float scale = 0.08838834764831845f;  // 1/sqrt(128)

  // Q A-fragments: 16 rows x 128 d = 4 chunks of 16x32, held in registers.
  v16h qa[4];
  {
    const _Float16* qrow = Qh + ((size_t)h * SEQ + q0 + wave * 16 + ln) * HDIM;
#pragma unroll
    for (int c = 0; c < 4; ++c) qa[c] = load_frag_split(qrow + c * 32, hi * 8);
  }

  v8f acc[8] = {};
  float mrow[8], lrow[8];
#pragma unroll
  for (int r = 0; r < 8; ++r) { mrow[r] = -1e30f; lrow[r] = 0.f; }

  for (int kt = 0; kt < SEQ; kt += 64) {
    __syncthreads();  // prior tile's LDS reads fully consumed
    {
      int key = tid >> 2, ch = (tid & 3) * 32;
      const _Float16* src = Kh + ((size_t)h * SEQ + kt + key) * HDIM + ch;
      _Float16* dst = &Kt[key][ch];
#pragma unroll
      for (int i = 0; i < 4; ++i) async_copy16(src + i * 8, dst + i * 8);

      int vkey = tid & 63;
      int db   = (tid >> 6) * 32;
      const _Float16* vsrc = Vh + ((size_t)h * SEQ + kt + vkey) * HDIM + db;
#pragma unroll
      for (int i = 0; i < 32; ++i) Vt[db + i][vkey] = vsrc[i];  // transpose into LDS

      if (kt + 64 < SEQ) {
        __builtin_prefetch(src + (size_t)64 * HDIM, 0, 1);
        __builtin_prefetch(vsrc + (size_t)64 * HDIM, 0, 1);
      }
    }
    async_wait();
    __syncthreads();

    // scores S = Q * K^T  (16 x 64 per wave): 4 d-chunks x 4 key-frags, batched loads
    v8f sc[4] = {};
#pragma unroll
    for (int c = 0; c < 4; ++c) {
      v16h b4[4];
#pragma unroll
      for (int nf = 0; nf < 4; ++nf)
        b4[nf] = load_frag_contig(&Kt[nf * 16 + ln][c * 32 + hi * 16]);
#pragma unroll
      for (int nf = 0; nf < 4; ++nf) sc[nf] = WMMA_F16(qa[c], b4[nf], sc[nf]);
    }

    // online softmax; stats per row (VGPR r <-> row r+8*hi), reduced across the 16-lane half
#pragma unroll
    for (int r = 0; r < 8; ++r) {
      float mx = -1e30f;
#pragma unroll
      for (int nf = 0; nf < 4; ++nf) { sc[nf][r] *= scale; mx = fmaxf(mx, sc[nf][r]); }
#pragma unroll
      for (int mask = 1; mask <= 8; mask <<= 1) mx = fmaxf(mx, __shfl_xor(mx, mask, 32));
      float mnew = fmaxf(mrow[r], mx);
      float cf   = __expf(mrow[r] - mnew);
      mrow[r] = mnew;
      float rs = 0.f;
#pragma unroll
      for (int nf = 0; nf < 4; ++nf) {
        float p = __expf(sc[nf][r] - mnew);
        sc[nf][r] = p;
        rs += p;
      }
#pragma unroll
      for (int mask = 1; mask <= 8; mask <<= 1) rs += __shfl_xor(rs, mask, 32);
      lrow[r] = lrow[r] * cf + rs;
#pragma unroll
      for (int nf = 0; nf < 8; ++nf) acc[nf][r] *= cf;
    }

    // P: C-layout frags -> per-wave LDS tile (LDS ops are in-order within a wave)
#pragma unroll
    for (int nf = 0; nf < 4; ++nf)
#pragma unroll
      for (int r = 0; r < 8; ++r)
        Pt[wave][hi * 8 + r][nf * 16 + ln] = (_Float16)sc[nf][r];

    // O += P * V : 2 key-chunks x 8 d-frags, batched loads (groups of 4)
#pragma unroll
    for (int kc = 0; kc < 2; ++kc) {
      v16h ap = load_frag_split(&Pt[wave][ln][kc * 32], hi * 8);
#pragma unroll
      for (int g2 = 0; g2 < 2; ++g2) {
        v16h bv[4];
#pragma unroll
        for (int j = 0; j < 4; ++j)
          bv[j] = load_frag_contig(&Vt[(g2 * 4 + j) * 16 + ln][kc * 32 + hi * 16]);
#pragma unroll
        for (int j = 0; j < 4; ++j)
          acc[g2 * 4 + j] = WMMA_F16(ap, bv[j], acc[g2 * 4 + j]);
      }
    }
  }

  float inv[8];
#pragma unroll
  for (int r = 0; r < 8; ++r) inv[r] = 1.0f / lrow[r];
#pragma unroll
  for (int nf = 0; nf < 8; ++nf)
#pragma unroll
    for (int r = 0; r < 8; ++r) {
      int row = q0 + wave * 16 + hi * 8 + r;
      Oh[(size_t)row * DMODEL + h * HDIM + nf * 16 + ln] = (_Float16)(acc[nf][r] * inv[r]);
    }
}

// ---------------------------------------------------------------- launch
extern "C" void kernel_launch(void* const* d_in, const int* in_sizes, int n_in,
                              void* d_out, int out_size, void* d_ws, size_t ws_size,
                              hipStream_t stream) {
  (void)in_sizes; (void)n_in; (void)out_size; (void)ws_size;
  const float* x   = (const float*)d_in[0];
  const float* wq  = (const float*)d_in[1];
  const float* wk  = (const float*)d_in[2];
  const float* wv  = (const float*)d_in[3];
  const float* wo  = (const float*)d_in[4];
  const float* qnw = (const float*)d_in[5];
  const float* knw = (const float*)d_in[6];
  float* out = (float*)d_out;

  char* ws = (char*)d_ws;
  size_t off = 0;
  auto alloc = [&](size_t bytes) -> char* {
    char* p = ws + off;
    off += (bytes + 255) & ~(size_t)255;
    return p;
  };
  _Float16* xh  = (_Float16*)alloc((size_t)SEQ * DMODEL * 2);       // 16 MB
  _Float16* wqh = (_Float16*)alloc((size_t)DMODEL * DMODEL * 2);    //  8 MB
  _Float16* wkh = (_Float16*)alloc((size_t)DMODEL * DMODEL * 2);
  _Float16* wvh = (_Float16*)alloc((size_t)DMODEL * DMODEL * 2);
  _Float16* woh = (_Float16*)alloc((size_t)DMODEL * DMODEL * 2);
  _Float16* Qh  = (_Float16*)alloc((size_t)NHEADS * SEQ * HDIM * 2);  // [H][S][128]
  _Float16* Kh  = (_Float16*)alloc((size_t)NHEADS * SEQ * HDIM * 2);
  _Float16* Vh  = (_Float16*)alloc((size_t)NHEADS * SEQ * HDIM * 2);
  _Float16* Oh  = (_Float16*)alloc((size_t)SEQ * DMODEL * 2);        // [S][H*128]

  cvt_f32_f16_kernel<<<2048, 256, 0, stream>>>(x,  xh,  SEQ * DMODEL);
  cvt_f32_f16_kernel<<<1024, 256, 0, stream>>>(wq, wqh, DMODEL * DMODEL);
  cvt_f32_f16_kernel<<<1024, 256, 0, stream>>>(wk, wkh, DMODEL * DMODEL);
  cvt_f32_f16_kernel<<<1024, 256, 0, stream>>>(wv, wvh, DMODEL * DMODEL);
  cvt_f32_f16_kernel<<<1024, 256, 0, stream>>>(wo, woh, DMODEL * DMODEL);

  dim3 gg(DMODEL / 128, SEQ / 128);
  // Q/K projections with fused per-head RMSNorm; V projection plain cast.
  gemm_kernel<<<gg, 256, 0, stream>>>(xh, wqh, nullptr, Qh, qnw, 1);
  gemm_kernel<<<gg, 256, 0, stream>>>(xh, wkh, nullptr, Kh, knw, 1);
  gemm_kernel<<<gg, 256, 0, stream>>>(xh, wvh, nullptr, Vh, nullptr, 1);

  flash_attn_kernel<<<dim3(SEQ / 128, NHEADS), 256, 0, stream>>>(Qh, Kh, Vh, Oh);

  // final projection -> f32 output
  gemm_kernel<<<gg, 256, 0, stream>>>(Oh, woh, out, nullptr, nullptr, 0);
}
